// IRFAggregator_39049842655549
// MI455X (gfx1250) — matrix operand order repaired and checked by reading, not compiled, not run on hardware
//
#include <hip/hip_runtime.h>
#include <math.h>

// ---------------------------------------------------------------------------
// IRFAggregator for MI455X (gfx1250): DFT-as-WMMA pipeline.
//   k_basis  : build irfft basis [160 x 128] pre-swizzled into WMMA B-fragment order
//   k_irf    : per-node gamma IRF -> forward DFT -> complex log (tables in ws, L2-resident)
//   k_coords : src/dst/diag int32 coords
//   k_paths  : per 16-row tile: gather+segment-sum logs, complex exp, stage f16 A-tile
//              in LDS (A-fragment swizzle), 40x v_wmma_f32_16x16x32_f16 irfft,
//              relu + cross-lane rowsum + flip + normalize + store
// ---------------------------------------------------------------------------

typedef __attribute__((ext_vector_type(16))) _Float16 v16h;
typedef __attribute__((ext_vector_type(8)))  float    v8f;

#define Nn    50000
#define Pp    200000
#define Tt    128
#define Ff    65          // rfft bins
#define KTOT  160         // 130 (re/im interleaved) padded to 5*32 for K=32 WMMA
#define ROWS  (Pp + Nn)   // 250000 = 16 * 15625 exactly
#define EPSF  1e-12f
#define PI2F  6.283185307179586f

__device__ __forceinline__ float softplusf(float x) {
    return (x > 20.f) ? x : log1pf(expf(x));
}

// Stage value into LDS in A-fragment order (16-bit A 16x32 layout, ISA 7.12.2):
// lane<16 : M=lane, K in {0..7,16..23}(+32*kb) ; lane>=16 : M=lane-16, K in {8..15,24..31}
__device__ __forceinline__ void stageA(_Float16* As, int j, int K, float v) {
    int kb = K >> 5, kk = K & 31;
    int hf = (kk >> 3) & 1;                  // lane-half
    int e  = (kk & 7) | ((kk & 16) >> 1);    // element 0..15 within v16h
    As[((kb * 2 + hf) * 16 + j) * 16 + e] = (_Float16)v;
}

// --------------------------- basis (B-fragment) ----------------------------
// B 32x16 f16 layout (matches A): lanes 0-15 are columns N=0..15 holding
// K in {0..7,16..23}; lanes 16-31 hold K in {8..15,24..31}.
// basis[kb][ct][lane][e] ; value = irfft weight for (K = kb*32 + 8*(lane>>4) + (e&7) + 16*(e>>3),
//                                                    t = ct*16 + (lane&15))
// x[t] = (1/128)[ReF0 + 2*sum_{k=1..63}(ReFk cos - ImFk sin) + ReF64*(-1)^t]
__global__ void k_basis(_Float16* __restrict__ basis) {
    int idx = blockIdx.x * blockDim.x + threadIdx.x;
    if (idx >= 5 * 8 * 32 * 16) return;
    int e    = idx & 15;
    int lane = (idx >> 4) & 31;
    int ct   = (idx >> 9) & 7;
    int kb   = idx >> 12;
    int t = ct * 16 + (lane & 15);
    int K = kb * 32 + ((lane >> 4) * 8) + (e & 7) + ((e >> 3) * 16);
    float val = 0.f;
    if (K < 2 * Ff) {
        int k = K >> 1;
        float ang = PI2F * (float)(k * t) / 128.f;
        if ((K & 1) == 0) {                       // Re coefficient
            float w = (k == 0 || k == 64) ? 1.f : 2.f;
            val = w * cosf(ang) * (1.f / 128.f);
        } else if (k != 0 && k != 64) {           // Im coefficient (bins 0,64 ignored)
            val = -2.f * sinf(ang) * (1.f / 128.f);
        }
    }
    basis[idx] = (_Float16)val;
}

// ---------------------- per-node IRF + DFT + log ---------------------------
__global__ void k_irf(const float* __restrict__ params,
                      float* __restrict__ logtab,   // [N][65][2]
                      float* __restrict__ freqtab)  // [N][65][2]
{
    __shared__ float sirf[Tt];
    int n = blockIdx.x;
    int t = threadIdx.x;
    float a = softplusf(params[2 * n])     + 1.0f;
    float b = softplusf(params[2 * n + 1]) + 0.5f;
    float tv = (float)t + 0.5f;
    float lg = (a - 1.f) * logf(tv) - tv / b - lgammaf(a) - a * logf(b);
    sirf[t] = expf(lg);                     // * DT(=1)
    __syncthreads();
    if (t < Ff) {
        float re = 0.f, im = 0.f;
        for (int j = 0; j < Tt; ++j) {
            float s, c;
            sincosf(-PI2F * (float)(t * j) / 128.f, &s, &c);
            re = fmaf(sirf[j], c, re);
            im = fmaf(sirf[j], s, im);
        }
        float rep = re + EPSF;              // log(F + EPS)
        float lr  = 0.5f * logf(rep * rep + im * im);
        float li  = atan2f(im, rep);
        size_t base = ((size_t)n * Ff + t) * 2;
        logtab[base]  = lr;  logtab[base + 1]  = li;
        freqtab[base] = re;  freqtab[base + 1] = im;
    }
}

// -------------------------------- coords -----------------------------------
__global__ void k_coords(const int* __restrict__ edges,
                         const int* __restrict__ pcs,
                         int* __restrict__ out) {
    int i = blockIdx.x * blockDim.x + threadIdx.x;
    if (i >= ROWS) return;
    int s, d;
    if (i < Pp) { s = edges[pcs[i]]; d = edges[pcs[i + 1] - 1]; }
    else        { s = i - Pp;        d = s; }
    out[i]        = s;
    out[ROWS + i] = d;
}

// ------------------- fused gather + exp + WMMA-irfft -----------------------
__global__ void __launch_bounds__(32)
k_paths(const int* __restrict__ edges, const int* __restrict__ pcs,
        const float* __restrict__ logtab, const float* __restrict__ freqtab,
        const _Float16* __restrict__ basis, float* __restrict__ out)
{
    __shared__ __align__(32) _Float16 As[5 * 2 * 16 * 16];  // 5.0 KB A-tile
    const int lane = threadIdx.x;
    const int r0   = blockIdx.x * 16;

    for (int i = lane; i < 5 * 2 * 16 * 16; i += 32) As[i] = (_Float16)0.0f;
    __syncthreads();

    // Stage 16 rows of the [rows x 160] frequency matrix (f16, A-fragment order)
    for (int j = 0; j < 16; ++j) {
        int row = r0 + j;
        if (row < Pp) {
            int eb = pcs[row], ee = pcs[row + 1];
            for (int k = lane; k < Ff; k += 32) {
                float lr = 0.f, lm = 0.f;
                for (int e = eb; e < ee; ++e) {           // 6 L2-resident gathers
                    size_t bi = ((size_t)edges[e] * Ff + k) * 2;
                    lr += logtab[bi]; lm += logtab[bi + 1];
                }
                float mag = expf(lr);
                float s, c; sincosf(lm, &s, &c);
                stageA(As, j, 2 * k,     mag * c);
                stageA(As, j, 2 * k + 1, mag * s);
            }
        } else {
            int nd = row - Pp;
            for (int k = lane; k < Ff; k += 32) {
                size_t bi = ((size_t)nd * Ff + k) * 2;
                stageA(As, j, 2 * k,     freqtab[bi]);
                stageA(As, j, 2 * k + 1, freqtab[bi + 1]);
            }
        }
    }
    __syncthreads();

    const int m  = lane & 15;
    const int hf = lane >> 4;

    v16h a[5];
#pragma unroll
    for (int kb = 0; kb < 5; ++kb)
        a[kb] = *(const v16h*)&As[((kb * 2 + hf) * 16 + m) * 16];

    v8f c[8];
#pragma unroll
    for (int ct = 0; ct < 8; ++ct) {
        v8f acc = {};
#pragma unroll
        for (int kb = 0; kb < 5; ++kb) {
            v16h b = *(const v16h*)(basis + (size_t)(((kb * 8 + ct) * 32 + lane) * 16));
            acc = __builtin_amdgcn_wmma_f32_16x16x32_f16(
                      false, a[kb], false, b, (short)0, acc, false, false);
        }
        c[ct] = acc;
    }

    // relu + per-row sum (C/D layout: VGPR j -> row j (lanes 0-15) / row j+8)
    float rs[8];
#pragma unroll
    for (int j = 0; j < 8; ++j) rs[j] = 0.f;
#pragma unroll
    for (int ct = 0; ct < 8; ++ct) {
#pragma unroll
        for (int j = 0; j < 8; ++j) {
            float v = c[ct][j];
            v = v > 0.f ? v : 0.f;
            c[ct][j] = v;
            rs[j] += v;
        }
    }
#pragma unroll
    for (int j = 0; j < 8; ++j) {           // reduce across the 16-lane half
        float v = rs[j];
        v += __shfl_xor(v, 1, 16);
        v += __shfl_xor(v, 2, 16);
        v += __shfl_xor(v, 4, 16);
        v += __shfl_xor(v, 8, 16);
        rs[j] = 1.f / (v + EPSF);
    }

    const int rowbase = r0 + hf * 8;
#pragma unroll
    for (int ct = 0; ct < 8; ++ct) {
        int t = ct * 16 + m;
#pragma unroll
        for (int j = 0; j < 8; ++j)          // flip along time + normalize
            out[(size_t)(rowbase + j) * Tt + (Tt - 1 - t)] = c[ct][j] * rs[j];
    }
}

// ---------------------------------------------------------------------------
extern "C" void kernel_launch(void* const* d_in, const int* in_sizes, int n_in,
                              void* d_out, int out_size, void* d_ws, size_t ws_size,
                              hipStream_t stream) {
    const float* params = (const float*)d_in[0];
    const int*   edges  = (const int*)d_in[1];
    const int*   pcs    = (const int*)d_in[2];

    // ws: logtab [N*65*2] f32 | freqtab [N*65*2] f32 | basis [5*8*32*16] f16
    float*    logtab  = (float*)d_ws;
    float*    freqtab = logtab + (size_t)Nn * Ff * 2;
    _Float16* basis   = (_Float16*)((char*)d_ws + 2ull * Nn * Ff * 2ull * sizeof(float));

    int*   coords = (int*)d_out;                 // [2 x ROWS] int32 bits
    float* agg    = (float*)d_out + 2 * ROWS;    // [ROWS x 128] f32

    k_basis <<<(5 * 8 * 32 * 16 + 255) / 256, 256, 0, stream>>>(basis);
    k_irf   <<<Nn, Tt, 0, stream>>>(params, logtab, freqtab);
    k_coords<<<(ROWS + 255) / 256, 256, 0, stream>>>(edges, pcs, coords);
    k_paths <<<ROWS / 16, 32, 0, stream>>>(edges, pcs, logtab, freqtab, basis, agg);
}
